// EGNNDynamicsTransferableMD_85349590106528
// MI455X (gfx1250) — matrix operand order repaired and checked by reading, not compiled, not run on hardware
//
#include <hip/hip_runtime.h>

#define BATCH   8
#define NPART   256
#define HDIM    64
#define NLAYERS 4
#define BN      (BATCH*NPART)
#define CRANGE  (15.0f/4.0f)

typedef __attribute__((ext_vector_type(16))) __bf16 v16bf;
typedef __attribute__((ext_vector_type(8)))  float  v8f;

__device__ __forceinline__ unsigned short f2bf(float f) {
  unsigned int u = __float_as_uint(f);
  u += 0x7FFFu + ((u >> 16) & 1u);           // round-to-nearest-even
  return (unsigned short)(u >> 16);
}
__device__ __forceinline__ float bf2f(unsigned short h) {
  return __uint_as_float(((unsigned int)h) << 16);
}
// fast activations: v_rcp_f32 instead of IEEE division chain
__device__ __forceinline__ float siluf(float x) {
  return x * __builtin_amdgcn_rcpf(1.0f + __expf(-x));
}
__device__ __forceinline__ float sigmf(float x) {
  return __builtin_amdgcn_rcpf(1.0f + __expf(-x));
}

// ---- A fragment: 16x32 bf16, CDNA5 layout ----
// lanes 0-15: M=lane, VGPR0-3 hold K=0..7 (pairs), VGPR4-7 hold K=16..23
// lanes16-31: M=lane-16, VGPR0-3 hold K=8..15,    VGPR4-7 hold K=24..31
__device__ __forceinline__ v16bf load_afrag(const unsigned short* ms, int kc) {
  int lane = threadIdx.x & 31;
  int m = lane & 15;
  int kb = ((lane < 16) ? 0 : 8) + kc * 32;
  union { v16bf v; unsigned int ui[8]; } f;
  const unsigned short* row = ms + m * HDIM;
#pragma unroll
  for (int i = 0; i < 8; ++i) {
    int k0 = kb + ((i < 4) ? (2 * i) : (16 + 2 * (i - 4)));
    f.ui[i] = *(const unsigned int*)(row + k0);   // two contiguous bf16
  }
  return f.v;
}

// ---- B fragment: 32x16 bf16, pre-swizzled in global: [slot][lane][16] ----
// lanes 0-15 hold K=0..7,16..23 (col N=lane), lanes 16-31 hold K=8..15,24..31
__device__ __forceinline__ v16bf load_bfrag(const unsigned short* swz, int slot) {
  int lane = threadIdx.x & 31;
  union { v16bf v; uint4 u4[2]; } f;
  const uint4* p = (const uint4*)(swz + (size_t)slot * 512 + lane * 16);
  f.u4[0] = p[0];
  f.u4[1] = p[1];
  return f.v;
}

// =================== setup kernels ===================

// h0 = embedding( [atom,aa,aa_pos,t]*nm ), x0 = x*nm, coord = x0
__global__ __launch_bounds__(64) void k_embed(
    const float* __restrict__ tB, const float* __restrict__ x,
    const float* __restrict__ at, const float* __restrict__ aat,
    const float* __restrict__ aap, const float* __restrict__ nmk,
    const float* __restrict__ embw, const float* __restrict__ embb,
    float* __restrict__ h0, float* __restrict__ x0, float* __restrict__ coord) {
  int g = blockIdx.x, o = threadIdx.x;
  int b = g >> 8;
  float nm = nmk[g];
  float f0 = at[g] * nm, f1 = aat[g] * nm, f2 = aap[g] * nm, f3 = tB[b] * nm;
  h0[g * HDIM + o] = f0 * embw[o * 4 + 0] + f1 * embw[o * 4 + 1] +
                     f2 * embw[o * 4 + 2] + f3 * embw[o * 4 + 3] + embb[o];
  if (o < 3) {
    float xv = x[g * 3 + o] * nm;
    x0[g * 3 + o] = xv;
    coord[g * 3 + o] = xv;
  }
}

// pre-swizzle edge_w2 and coord_w1 into WMMA-B fragment order (bf16)
__global__ void k_swz(const float* __restrict__ ew2, const float* __restrict__ cw1,
                      unsigned short* __restrict__ w2s, unsigned short* __restrict__ wc1s) {
  for (int idx = threadIdx.x + blockIdx.x * blockDim.x;
       idx < NLAYERS * 4 * 2 * 32 * 16; idx += blockDim.x * gridDim.x) {
    int j    = idx & 15;
    int lane = (idx >> 4) & 31;
    int kc   = (idx >> 9) & 1;
    int nt   = (idx >> 10) & 3;
    int l    = idx >> 12;
    int n = nt * 16 + (lane & 15);
    int k = kc * 32 + ((lane < 16) ? 0 : 8) + ((j < 8) ? j : (8 + j));
    w2s[idx]  = f2bf(ew2[(l * HDIM + n) * HDIM + k]);
    wc1s[idx] = f2bf(cw1[(l * HDIM + n) * HDIM + k]);
  }
}

// per-node halves of the edge layer-1 GEMM: rowpart = h@W1[:, :64]^T, colpart = h@W1[:, 64:128]^T
__global__ __launch_bounds__(64) void k_part(const float* __restrict__ h,
                                             const float* __restrict__ ew1, int l,
                                             float* __restrict__ rowp, float* __restrict__ colp) {
  int g = blockIdx.x, o = threadIdx.x;
  __shared__ float hl[HDIM];
  hl[o] = h[g * HDIM + o];
  __syncthreads();
  const float* W = ew1 + (size_t)(l * HDIM + o) * 130;
  float rs = 0.f, cs = 0.f;
#pragma unroll 8
  for (int k = 0; k < HDIM; ++k) { rs += hl[k] * W[k]; cs += hl[k] * W[64 + k]; }
  rowp[g * HDIM + o] = rs;
  colp[g * HDIM + o] = cs;
}

// =================== per-layer edge kernel (WMMA) ===================

__global__ __launch_bounds__(256) void k_edge(
    const float* __restrict__ coord_in, const float* __restrict__ x0g,
    const float* __restrict__ nmk,
    const float* __restrict__ rowp, const float* __restrict__ colp,
    const float* __restrict__ ew1, const float* __restrict__ eb1,
    const unsigned short* __restrict__ w2s, const float* __restrict__ eb2,
    const unsigned short* __restrict__ wc1s, const float* __restrict__ cb1,
    const float* __restrict__ attw, const float* __restrict__ attbv,
    const float* __restrict__ cw2, int l,
    float* __restrict__ coord_out, float* __restrict__ agg_out) {

  __shared__ unsigned short s_cpart[NPART][HDIM];    // 32768 B  colpart (bf16)
  __shared__ unsigned short s_mstage[8][16][HDIM];   // 16384 B  per-wave A staging
  __shared__ float s_xc[NPART][4];                   //  4096 B  cur coords + nm
  __shared__ float s_x0c[NPART][3];                  //  3072 B  initial coords
  __shared__ float s_vec[8][HDIM];                   //  2048 B  rp,b1,w1r,w1e,b2,bc1,attw,cw2
  __shared__ float s_wsc[8][16][8];                  //  4096 B  rad,ea0,em,nd xyz per edge
  __shared__ float s_aggp[8][HDIM];                  //  2048 B  per-wave agg partials
  __shared__ float s_caccp[8][4];                    //   128 B  per-wave coord partials

  int g = blockIdx.x;
  int b = g >> 8;
  int iloc = g & 255;
  int tid = threadIdx.x;
  int lane = tid & 31;
  int w = tid >> 5;
  int half = lane >> 4;
  int n = lane & 15;

  { // phase 0: cooperative loads
    int j = tid;
    int gj = b * NPART + j;
    s_xc[j][0] = coord_in[gj * 3 + 0];
    s_xc[j][1] = coord_in[gj * 3 + 1];
    s_xc[j][2] = coord_in[gj * 3 + 2];
    s_xc[j][3] = nmk[gj];
    s_x0c[j][0] = x0g[gj * 3 + 0];
    s_x0c[j][1] = x0g[gj * 3 + 1];
    s_x0c[j][2] = x0g[gj * 3 + 2];
#pragma unroll 4
    for (int ch = 0; ch < HDIM; ++ch) s_cpart[j][ch] = f2bf(colp[gj * HDIM + ch]);
    if (tid < HDIM) {
      int ch = tid;
      s_vec[0][ch] = rowp[g * HDIM + ch];
      s_vec[1][ch] = eb1[l * HDIM + ch];
      s_vec[2][ch] = ew1[(size_t)(l * HDIM + ch) * 130 + 128];
      s_vec[3][ch] = ew1[(size_t)(l * HDIM + ch) * 130 + 129];
      s_vec[4][ch] = eb2[l * HDIM + ch];
      s_vec[5][ch] = cb1[l * HDIM + ch];
      s_vec[6][ch] = attw[l * HDIM + ch];
      s_vec[7][ch] = cw2[l * HDIM + ch];
    }
  }
  float attb = attbv[l];
  __syncthreads();

  float aggreg[4] = {0.f, 0.f, 0.f, 0.f};
  float cax = 0.f, cay = 0.f, caz = 0.f;

  for (int tt = 0; tt < 2; ++tt) {
    int j0 = (w + tt * 8) * 16;

    // ---- step 1: per-edge scalars ----
    if (lane < 16) {
      int m = lane, j = j0 + m;
      float dx = s_xc[iloc][0] - s_xc[j][0];
      float dy = s_xc[iloc][1] - s_xc[j][1];
      float dz = s_xc[iloc][2] - s_xc[j][2];
      float rad = dx * dx + dy * dy + dz * dz;
      float inv = __builtin_amdgcn_rcpf(sqrtf(rad + 1e-8f) + 1.0f);
      float ex = s_x0c[iloc][0] - s_x0c[j][0];
      float ey = s_x0c[iloc][1] - s_x0c[j][1];
      float ez = s_x0c[iloc][2] - s_x0c[j][2];
      float ea = ex * ex + ey * ey + ez * ez;
      float em = s_xc[iloc][3] * s_xc[j][3] * ((j == iloc) ? 0.0f : 1.0f);
      s_wsc[w][m][0] = rad; s_wsc[w][m][1] = ea; s_wsc[w][m][2] = em;
      s_wsc[w][m][3] = dx * inv; s_wsc[w][m][4] = dy * inv; s_wsc[w][m][5] = dz * inv;
    }
    __syncthreads();

    // ---- step 2: m1 = silu(rowpart + colpart + radial*w1r + ea0*w1e + b1), stage bf16 ----
    for (int idx = lane; idx < 16 * HDIM; idx += 32) {
      int m = idx >> 6, ch = idx & 63;
      float v = s_vec[0][ch] + bf2f(s_cpart[j0 + m][ch]) +
                s_wsc[w][m][0] * s_vec[2][ch] + s_wsc[w][m][1] * s_vec[3][ch] +
                s_vec[1][ch];
      s_mstage[w][m][ch] = f2bf(siluf(v));
    }
    __syncthreads();

    // ---- step 3: m2 = silu(m1 @ W2^T + b2)  (WMMA) ----
    float m2v[4][8];
    {
      v16bf a0 = load_afrag(&s_mstage[w][0][0], 0);
      v16bf a1 = load_afrag(&s_mstage[w][0][0], 1);
#pragma unroll
      for (int nt = 0; nt < 4; ++nt) {
        v8f acc = {0.f, 0.f, 0.f, 0.f, 0.f, 0.f, 0.f, 0.f};
        v16bf wb0 = load_bfrag(w2s, (l * 4 + nt) * 2 + 0);
        v16bf wb1 = load_bfrag(w2s, (l * 4 + nt) * 2 + 1);
        acc = __builtin_amdgcn_wmma_f32_16x16x32_bf16(false, a0, false, wb0, (short)0, acc, false, false);
        acc = __builtin_amdgcn_wmma_f32_16x16x32_bf16(false, a1, false, wb1, (short)0, acc, false, false);
        float bias = s_vec[4][nt * 16 + n];
#pragma unroll
        for (int r = 0; r < 8; ++r) m2v[nt][r] = siluf(acc[r] + bias);
      }
    }

    // ---- step 4: attention dot over 64 channels (shuffle butterfly per 16-lane half) ----
    float sig[8];
    {
      float part[8];
#pragma unroll
      for (int r = 0; r < 8; ++r)
        part[r] = m2v[0][r] * s_vec[6][n] + m2v[1][r] * s_vec[6][16 + n] +
                  m2v[2][r] * s_vec[6][32 + n] + m2v[3][r] * s_vec[6][48 + n];
#pragma unroll
      for (int mask = 1; mask < 16; mask <<= 1)
#pragma unroll
        for (int r = 0; r < 8; ++r) part[r] += __shfl_xor(part[r], mask, 32);
#pragma unroll
      for (int r = 0; r < 8; ++r) sig[r] = sigmf(part[r] + attb);
    }

    // ---- step 5: gate by sigmoid * edge-mask, restage bf16, per-wave agg partial ----
    {
      float emr[8];
#pragma unroll
      for (int r = 0; r < 8; ++r) emr[r] = s_wsc[w][r + half * 8][2];
#pragma unroll
      for (int nt = 0; nt < 4; ++nt) {
        float cs = 0.f;
#pragma unroll
        for (int r = 0; r < 8; ++r) {
          float mv = m2v[nt][r] * sig[r] * emr[r];
          s_mstage[w][r + half * 8][nt * 16 + n] = f2bf(mv);
          cs += mv;
        }
        cs += __shfl_xor(cs, 16, 32);   // combine row halves (same channel)
        if (half == 0) aggreg[nt] += cs;
      }
    }
    __syncthreads();

    // ---- step 6: coord hidden = silu(m_att @ Wc1^T + bc1)  (WMMA) ----
    float ccv[4][8];
    {
      v16bf a0 = load_afrag(&s_mstage[w][0][0], 0);
      v16bf a1 = load_afrag(&s_mstage[w][0][0], 1);
#pragma unroll
      for (int nt = 0; nt < 4; ++nt) {
        v8f acc = {0.f, 0.f, 0.f, 0.f, 0.f, 0.f, 0.f, 0.f};
        v16bf wb0 = load_bfrag(wc1s, (l * 4 + nt) * 2 + 0);
        v16bf wb1 = load_bfrag(wc1s, (l * 4 + nt) * 2 + 1);
        acc = __builtin_amdgcn_wmma_f32_16x16x32_bf16(false, a0, false, wb0, (short)0, acc, false, false);
        acc = __builtin_amdgcn_wmma_f32_16x16x32_bf16(false, a1, false, wb1, (short)0, acc, false, false);
        float bias = s_vec[5][nt * 16 + n];
#pragma unroll
        for (int r = 0; r < 8; ++r) ccv[nt][r] = siluf(acc[r] + bias);
      }
    }

    // ---- step 7: phi = tanh(hidden @ cw2); accumulate trans over tile ----
    {
      float part[8];
#pragma unroll
      for (int r = 0; r < 8; ++r)
        part[r] = ccv[0][r] * s_vec[7][n] + ccv[1][r] * s_vec[7][16 + n] +
                  ccv[2][r] * s_vec[7][32 + n] + ccv[3][r] * s_vec[7][48 + n];
#pragma unroll
      for (int mask = 1; mask < 16; mask <<= 1)
#pragma unroll
        for (int r = 0; r < 8; ++r) part[r] += __shfl_xor(part[r], mask, 32);
      float sx = 0.f, sy = 0.f, sz = 0.f;
#pragma unroll
      for (int r = 0; r < 8; ++r) {
        int row = r + half * 8;
        float p = tanhf(part[r]) * s_wsc[w][row][2];   // * em
        sx += p * s_wsc[w][row][3];
        sy += p * s_wsc[w][row][4];
        sz += p * s_wsc[w][row][5];
      }
      sx += __shfl_xor(sx, 16, 32);
      sy += __shfl_xor(sy, 16, 32);
      sz += __shfl_xor(sz, 16, 32);
      if (lane == 0) { cax += sx * CRANGE; cay += sy * CRANGE; caz += sz * CRANGE; }
    }
    __syncthreads();
  }

  // deterministic block reduction of per-wave partials
  if (half == 0) {
#pragma unroll
    for (int nt = 0; nt < 4; ++nt) s_aggp[w][nt * 16 + n] = aggreg[nt];
  }
  if (lane == 0) { s_caccp[w][0] = cax; s_caccp[w][1] = cay; s_caccp[w][2] = caz; }
  __syncthreads();

  float nm_i = s_xc[iloc][3];
  if (tid < HDIM) {
    float a = 0.f;
#pragma unroll
    for (int ww = 0; ww < 8; ++ww) a += s_aggp[ww][tid];
    agg_out[g * HDIM + tid] = a;
  }
  if (tid < 3) {
    float c = 0.f;
#pragma unroll
    for (int ww = 0; ww < 8; ++ww) c += s_caccp[ww][tid];
    coord_out[g * 3 + tid] = (s_xc[iloc][tid] + c) * nm_i;
  }
}

// =================== per-layer node update ===================

__global__ __launch_bounds__(64) void k_node(
    const float* __restrict__ h, const float* __restrict__ agg,
    const float* __restrict__ nw1, const float* __restrict__ nb1,
    const float* __restrict__ nw2, const float* __restrict__ nb2,
    const float* __restrict__ nmk, int l, float* __restrict__ hout) {
  int g = blockIdx.x, o = threadIdx.x;
  __shared__ float hl[HDIM], al[HDIM], t1[HDIM];
  hl[o] = h[g * HDIM + o];
  al[o] = agg[g * HDIM + o];
  __syncthreads();
  const float* W1 = nw1 + (size_t)(l * HDIM + o) * 128;
  float s = nb1[l * HDIM + o];
#pragma unroll 8
  for (int k = 0; k < HDIM; ++k) s += hl[k] * W1[k] + al[k] * W1[64 + k];
  t1[o] = siluf(s);
  __syncthreads();
  const float* W2 = nw2 + (size_t)(l * HDIM + o) * HDIM;
  float u = nb2[l * HDIM + o];
#pragma unroll 8
  for (int k = 0; k < HDIM; ++k) u += t1[k] * W2[k];
  hout[g * HDIM + o] = (hl[o] + u) * nmk[g];
}

// =================== finalize: vel = coord - x0, remove masked mean ===================

__global__ __launch_bounds__(256) void k_final(
    const float* __restrict__ cf, const float* __restrict__ x0g,
    const float* __restrict__ nmk, float* __restrict__ out) {
  __shared__ float red[NPART][4];
  int b = blockIdx.x, t = threadIdx.x;
  int g = b * NPART + t;
  float nm = nmk[g];
  float v0 = (cf[g * 3 + 0] - x0g[g * 3 + 0]) * nm;
  float v1 = (cf[g * 3 + 1] - x0g[g * 3 + 1]) * nm;
  float v2 = (cf[g * 3 + 2] - x0g[g * 3 + 2]) * nm;
  red[t][0] = v0; red[t][1] = v1; red[t][2] = v2; red[t][3] = nm;
  __syncthreads();
  for (int s = 128; s > 0; s >>= 1) {
    if (t < s) {
      red[t][0] += red[t + s][0]; red[t][1] += red[t + s][1];
      red[t][2] += red[t + s][2]; red[t][3] += red[t + s][3];
    }
    __syncthreads();
  }
  float inv = 1.0f / red[0][3];
  out[g * 3 + 0] = v0 - red[0][0] * inv * nm;
  out[g * 3 + 1] = v1 - red[0][1] * inv * nm;
  out[g * 3 + 2] = v2 - red[0][2] * inv * nm;
}

// =================== host launcher ===================

extern "C" void kernel_launch(void* const* d_in, const int* in_sizes, int n_in,
                              void* d_out, int out_size, void* d_ws, size_t ws_size,
                              hipStream_t stream) {
  const float* p_t    = (const float*)d_in[0];
  const float* p_x    = (const float*)d_in[1];
  const float* p_at   = (const float*)d_in[2];
  const float* p_aat  = (const float*)d_in[3];
  const float* p_aap  = (const float*)d_in[4];
  const float* p_nm   = (const float*)d_in[5];
  const float* p_embw = (const float*)d_in[6];
  const float* p_embb = (const float*)d_in[7];
  const float* p_ew1  = (const float*)d_in[8];
  const float* p_eb1  = (const float*)d_in[9];
  const float* p_ew2  = (const float*)d_in[10];
  const float* p_eb2  = (const float*)d_in[11];
  const float* p_nw1  = (const float*)d_in[12];
  const float* p_nb1  = (const float*)d_in[13];
  const float* p_nw2  = (const float*)d_in[14];
  const float* p_nb2  = (const float*)d_in[15];
  const float* p_cw1  = (const float*)d_in[16];
  const float* p_cb1  = (const float*)d_in[17];
  const float* p_attw = (const float*)d_in[18];
  const float* p_attb = (const float*)d_in[19];
  const float* p_cw2  = (const float*)d_in[20];

  char* ws = (char*)d_ws;
  float* x0b    = (float*)(ws + 0);                 // 24576 B
  float* coordA = (float*)(ws + 32768);
  float* coordB = (float*)(ws + 65536);
  float* hA     = (float*)(ws + 98304);             // 524288 B
  float* hB     = (float*)(ws + 622592);
  float* rowp   = (float*)(ws + 1146880);
  float* colp   = (float*)(ws + 1671168);
  float* aggb   = (float*)(ws + 2195456);
  unsigned short* w2s  = (unsigned short*)(ws + 2719744);  // 32768 B
  unsigned short* wc1s = (unsigned short*)(ws + 2752512);  // 32768 B

  k_embed<<<BN, 64, 0, stream>>>(p_t, p_x, p_at, p_aat, p_aap, p_nm,
                                 p_embw, p_embb, hA, x0b, coordA);
  k_swz<<<16, 256, 0, stream>>>(p_ew2, p_cw1, w2s, wc1s);

  float* hcur = hA;  float* hnxt = hB;
  float* ccur = coordA; float* cnxt = coordB;
  for (int l = 0; l < NLAYERS; ++l) {
    k_part<<<BN, 64, 0, stream>>>(hcur, p_ew1, l, rowp, colp);
    k_edge<<<BN, 256, 0, stream>>>(ccur, x0b, p_nm, rowp, colp,
                                   p_ew1, p_eb1, w2s, p_eb2, wc1s, p_cb1,
                                   p_attw, p_attb, p_cw2, l, cnxt, aggb);
    k_node<<<BN, 64, 0, stream>>>(hcur, aggb, p_nw1, p_nb1, p_nw2, p_nb2,
                                  p_nm, l, hnxt);
    float* tmp;
    tmp = hcur; hcur = hnxt; hnxt = tmp;
    tmp = ccur; ccur = cnxt; cnxt = tmp;
  }

  k_final<<<BATCH, 256, 0, stream>>>(ccur, x0b, p_nm, (float*)d_out);
}
